// HMM_22101901705281
// MI455X (gfx1250) — matrix operand order repaired and checked by reading, not compiled, not run on hardware
//
#include <hip/hip_runtime.h>

#define N_STATES 512
#define M_SYMS   4096
#define BATCH    64
#define TSEQ     512
#define BT       16          // batch rows per workgroup

typedef __attribute__((ext_vector_type(16))) __bf16 v16bf;
typedef __attribute__((ext_vector_type(8)))  float  v8f;

__device__ __forceinline__ unsigned short f2bf(float f) {
  unsigned int u = __float_as_uint(f);
  u += 0x7FFFu + ((u >> 16) & 1u);            // round-to-nearest-even
  return (unsigned short)(u >> 16);
}
__device__ __forceinline__ float bf2f(unsigned short h) {
  return __uint_as_float(((unsigned int)h) << 16);
}
__device__ __forceinline__ v16bf ld_frag(const unsigned short* p) {
  union { uint4 q[2]; v16bf v; } u;
  u.q[0] = *(const uint4*)(p);
  u.q[1] = *(const uint4*)(p + 8);
  return u.v;
}

// ---- preprocessing: emission row log-softmax, written transposed (M,N) ----
__global__ __launch_bounds__(256) void k_em(const float* __restrict__ em,
                                            float* __restrict__ emT) {
  __shared__ float red[8];
  const int j = blockIdx.x, tid = threadIdx.x, lane = tid & 31, wv = tid >> 5;
  const float* row = em + (size_t)j * M_SYMS;
  float mx = -INFINITY;
  for (int m = tid; m < M_SYMS; m += 256) mx = fmaxf(mx, row[m]);
  #pragma unroll
  for (int o = 16; o; o >>= 1) mx = fmaxf(mx, __shfl_xor(mx, o, 32));
  if (lane == 0) red[wv] = mx;
  __syncthreads();
  float bmx = red[0];
  for (int i = 1; i < 8; ++i) bmx = fmaxf(bmx, red[i]);
  float s = 0.f;
  for (int m = tid; m < M_SYMS; m += 256) s += __expf(row[m] - bmx);
  #pragma unroll
  for (int o = 16; o; o >>= 1) s += __shfl_xor(s, o, 32);
  __syncthreads();
  if (lane == 0) red[wv] = s;
  __syncthreads();
  float bs = 0.f;
  for (int i = 0; i < 8; ++i) bs += red[i];
  const float lse = bmx + __logf(bs);
  for (int m = tid; m < M_SYMS; m += 256)
    emT[(size_t)m * N_STATES + j] = row[m] - lse;
}

// ---- transition column logsumexp ----
__global__ __launch_bounds__(512) void k_collse(const float* __restrict__ tr,
                                                float* __restrict__ colLse) {
  const int k = threadIdx.x;
  float mx = -INFINITY;
  for (int j = 0; j < N_STATES; ++j) mx = fmaxf(mx, tr[(size_t)j * N_STATES + k]);
  float s = 0.f;
  for (int j = 0; j < N_STATES; ++j) s += __expf(tr[(size_t)j * N_STATES + k] - mx);
  colLse[k] = mx + __logf(s);
}

// ---- W[k,j] = exp(log_tr[j,k]) as bf16 hi/lo in WMMA B-fragment layout ----
__global__ __launch_bounds__(256) void k_wfrag(const float* __restrict__ tr,
                                               const float* __restrict__ colLse,
                                               unsigned short* __restrict__ Whi,
                                               unsigned short* __restrict__ Wlo) {
  const int e = blockIdx.x * 256 + threadIdx.x;   // over N*N
  const int k = e & (N_STATES - 1);
  const int j = e >> 9;
  const float wv = __expf(tr[(size_t)j * N_STATES + k] - colLse[k]);
  const unsigned short hi = f2bf(wv);
  const unsigned short lo = f2bf(wv - bf2f(hi));
  // B 32x16 bf16 fragment: lane = n + 16*(kk>>4), elem i = kk&15
  const int jt = j >> 4, nn = j & 15;
  const int kc = k >> 5, kk = k & 31;
  const int idx = (((jt * 16 + kc) * 32) + (nn + ((kk >> 4) << 4))) * 16 + (kk & 15);
  Whi[idx] = hi;
  Wlo[idx] = lo;
}

// ---- prior log-softmax ----
__global__ __launch_bounds__(512) void k_pri(const float* __restrict__ pri,
                                             float* __restrict__ logpri) {
  __shared__ float red[16];
  const int tid = threadIdx.x, lane = tid & 31, wv = tid >> 5;
  const float v = pri[tid];
  float mx = v;
  #pragma unroll
  for (int o = 16; o; o >>= 1) mx = fmaxf(mx, __shfl_xor(mx, o, 32));
  if (lane == 0) red[wv] = mx;
  __syncthreads();
  float bmx = red[0];
  for (int i = 1; i < 16; ++i) bmx = fmaxf(bmx, red[i]);
  float s = __expf(v - bmx);
  #pragma unroll
  for (int o = 16; o; o >>= 1) s += __shfl_xor(s, o, 32);
  __syncthreads();
  if (lane == 0) red[wv] = s;
  __syncthreads();
  float bs = 0.f;
  for (int i = 0; i < 16; ++i) bs += red[i];
  logpri[tid] = v - (bmx + __logf(bs));
}

// ---- main forward scan: 4 blocks x 512 threads, 16 batch rows each ----
__global__ __launch_bounds__(512) void hmm_fwd(
    const int* __restrict__ x, const int* __restrict__ Tlen,
    const float* __restrict__ emT, const unsigned short* __restrict__ Whi,
    const unsigned short* __restrict__ Wlo, const float* __restrict__ logpri,
    float* __restrict__ out) {
  __shared__ float alpha[BT][N_STATES];                    // centered log-alpha
  __shared__ __align__(16) unsigned short aHi[8][32][16];  // A frags, half-K
  __shared__ __align__(16) unsigned short aLo[8][32][16];
  __shared__ float rowmax[BT];
  __shared__ int   xrow[BT];

  const int tid  = threadIdx.x;
  const int lane = tid & 31;
  const int w    = tid >> 5;            // wave id == batch row for reductions
  const int b0   = blockIdx.x * BT;
  const int Tw   = Tlen[b0 + w];
  const int jt0  = 2 * w, jt1 = 2 * w + 1;
  const int n    = lane & 15;
  const int mAdd = (lane >> 4) << 3;    // C/D layout: lanes 16-31 hold M=8..15
  float Coff = 0.0f;                    // running per-row offset (alpha = stored + Coff)

  // alpha(0)
  for (int e = tid; e < BT * N_STATES; e += 512) {
    const int m = e >> 9, j = e & (N_STATES - 1);
    const int xv = x[(b0 + m) * TSEQ];
    alpha[m][j] = emT[(size_t)xv * N_STATES + j] + logpri[j];
  }

  for (int t = 0; t < TSEQ; ++t) {
    __syncthreads();
    // P1: wave w reduces row w (max + sumexp); emit output at t == T-1
    float mx = -INFINITY;
    for (int k = lane; k < N_STATES; k += 32) mx = fmaxf(mx, alpha[w][k]);
    #pragma unroll
    for (int o = 16; o; o >>= 1) mx = fmaxf(mx, __shfl_xor(mx, o, 32));
    float sm = 0.0f;
    for (int k = lane; k < N_STATES; k += 32) sm += __expf(alpha[w][k] - mx);
    #pragma unroll
    for (int o = 16; o; o >>= 1) sm += __shfl_xor(sm, o, 32);
    if (lane == 0) {
      rowmax[w] = mx;
      if (Tw - 1 == t) out[b0 + w] = Coff + mx + __logf(sm);
    }
    Coff += mx;
    if (t == TSEQ - 1) break;
    __syncthreads();
    if (tid < BT) xrow[tid] = x[(b0 + tid) * TSEQ + t + 1];

    v8f acc0 = {}, acc1 = {};
    #pragma unroll 1
    for (int half = 0; half < 2; ++half) {
      // build A hi/lo fragments for K range [half*256, half*256+256)
      const int kBase = half * 256;
      for (int e = tid; e < BT * 256; e += 512) {
        const int m = e >> 8;
        const int k = kBase + (e & 255);
        const float p = __expf(alpha[m][k] - rowmax[m]);
        const unsigned short hi = f2bf(p);
        const unsigned short lo = f2bf(p - bf2f(hi));
        const int kk = k & 31;
        int h, i;                         // A 16x32 bf16 fragment mapping
        h = kk >> 4; i = kk & 15;
        const int c = (k >> 5) & 7;
        const int ln = m + (h << 4);
        aHi[c][ln][i] = hi;
        aLo[c][ln][i] = lo;
      }
      __syncthreads();
      // Keep this loop ROLLED: full unroll previously forced scratch spills
      // (VGPR pressure > 250). Rolled body needs ~60 VGPRs: ah/al + bh/bl
      // (reused across both j-tiles) + two f32 accumulators.
      #pragma unroll 1
      for (int kc = 0; kc < 8; ++kc) {
        const v16bf ah = ld_frag(&aHi[kc][lane][0]);
        const v16bf al = ld_frag(&aLo[kc][lane][0]);
        const int kcg = half * 8 + kc;
        const int o0 = ((jt0 * 16 + kcg) * 32 + lane) << 4;
        const int o1 = ((jt1 * 16 + kcg) * 32 + lane) << 4;
        v16bf bh = ld_frag(Whi + o0);
        v16bf bl = ld_frag(Wlo + o0);
        acc0 = __builtin_amdgcn_wmma_f32_16x16x32_bf16(false, ah, false, bh, (short)0, acc0, false, false);
        acc0 = __builtin_amdgcn_wmma_f32_16x16x32_bf16(false, ah, false, bl, (short)0, acc0, false, false);
        acc0 = __builtin_amdgcn_wmma_f32_16x16x32_bf16(false, al, false, bh, (short)0, acc0, false, false);
        bh = ld_frag(Whi + o1);
        bl = ld_frag(Wlo + o1);
        acc1 = __builtin_amdgcn_wmma_f32_16x16x32_bf16(false, ah, false, bh, (short)0, acc1, false, false);
        acc1 = __builtin_amdgcn_wmma_f32_16x16x32_bf16(false, ah, false, bl, (short)0, acc1, false, false);
        acc1 = __builtin_amdgcn_wmma_f32_16x16x32_bf16(false, al, false, bh, (short)0, acc1, false, false);
      }
      __syncthreads();
    }
    // epilogue: alpha(t+1)[m][j] = obs(t+1) + log(s)   (stays centered)
    #pragma unroll
    for (int r = 0; r < 8; ++r) {
      const int m = r + mAdd;
      const int xv = xrow[m];
      alpha[m][jt0 * 16 + n] = emT[(size_t)xv * N_STATES + jt0 * 16 + n] + __logf(acc0[r]);
      alpha[m][jt1 * 16 + n] = emT[(size_t)xv * N_STATES + jt1 * 16 + n] + __logf(acc1[r]);
    }
  }
}

extern "C" void kernel_launch(void* const* d_in, const int* in_sizes, int n_in,
                              void* d_out, int out_size, void* d_ws, size_t ws_size,
                              hipStream_t stream) {
  (void)in_sizes; (void)n_in; (void)out_size; (void)ws_size;
  const int*   x   = (const int*)d_in[0];
  const int*   T   = (const int*)d_in[1];
  const float* em  = (const float*)d_in[2];
  const float* tr  = (const float*)d_in[3];
  const float* pri = (const float*)d_in[4];
  float* out = (float*)d_out;

  char* ws = (char*)d_ws;
  float*          emT    = (float*)ws;                                   // 8 MB
  unsigned short* Whi    = (unsigned short*)(ws + 8388608);              // 512 KB
  unsigned short* Wlo    = (unsigned short*)(ws + 8388608 + 524288);     // 512 KB
  float*          colLse = (float*)(ws + 8388608 + 1048576);             // 2 KB
  float*          logpri = colLse + N_STATES;                            // 2 KB

  k_em    <<<N_STATES, 256, 0, stream>>>(em, emT);
  k_collse<<<1, N_STATES, 0, stream>>>(tr, colLse);
  k_wfrag <<<(N_STATES * N_STATES) / 256, 256, 0, stream>>>(tr, colLse, Whi, Wlo);
  k_pri   <<<1, N_STATES, 0, stream>>>(pri, logpri);
  hmm_fwd <<<BATCH / BT, 512, 0, stream>>>(x, T, emT, Whi, Wlo, logpri, out);
}
